// TubeViT_v2_flowv4_54425825574965
// MI455X (gfx1250) — matrix operand order repaired and checked by reading, not compiled, not run on hardware
//
#include <hip/hip_runtime.h>
#include <cstdint>
#include <cstddef>

typedef __bf16 bf16_t;
typedef __attribute__((ext_vector_type(16))) __bf16 v16bf;
typedef __attribute__((ext_vector_type(8)))  float  v8f;

static inline unsigned divup_u(long long a, long long b) { return (unsigned)((a + b - 1) / b); }

// ---------------------------------------------------------------------------
// helpers
// ---------------------------------------------------------------------------
__device__ __forceinline__ bf16_t f2bf(float f) {
  // gfx1250 has native BF16 converts; let the backend pick.
  return (bf16_t)f;
}

#if __has_builtin(__builtin_amdgcn_global_load_async_to_lds_b128) && \
    __has_builtin(__builtin_amdgcn_s_wait_asynccnt)
#define ASYNC_LDS 1
typedef int native_i4 __attribute__((vector_size(16)));            // clang-native int4
typedef __attribute__((address_space(1))) native_i4 gas_i4;        // global int4
typedef __attribute__((address_space(3))) native_i4 las_i4;        // LDS int4
#else
#define ASYNC_LDS 0
#endif

#define LDSLD 40  // 32 + 8 bf16 pad per LDS row

// A-matrix fragment (16x32 bf16): lanes 0-15 rows hold K {0..7,16..23},
// lanes 16-31 hold K {8..15,24..31}.  p = row base in LDS, klo = 0 or 8.
__device__ __forceinline__ v16bf lds_fragA(const bf16_t* p, int klo) {
  union { v16bf v; uint4 q[2]; } u;
  u.q[0] = *(const uint4*)(p + klo);
  u.q[1] = *(const uint4*)(p + klo + 16);
  return u.v;
}
// B-matrix fragment (32x16 bf16): lane = column; lanes 0-15 hold K {0..7,16..23},
// lanes 16-31 hold K {8..15,24..31}.  p = column base, koff = 0 or 8.
__device__ __forceinline__ v16bf lds_fragB(const bf16_t* p, int koff) {
  union { v16bf v; uint4 q[2]; } u;
  u.q[0] = *(const uint4*)(p + koff);
  u.q[1] = *(const uint4*)(p + koff + 16);
  return u.v;
}

// ---------------------------------------------------------------------------
// Generic bf16 WMMA GEMM:  D[M,N] = A[M,K] * B(op) + bias + residual
// ---------------------------------------------------------------------------
constexpr int OPB_N  = 0, OPB_T  = 1, OPB_C9  = 2;   // B layouts
constexpr int BI_NONE = 0, BI_ROW = 1, BI_COL = 2;   // bias modes
constexpr int RE_NONE = 0, RE_NORM = 1, RE_TRANS = 2;
constexpr int STM_NORM = 0, STM_TRANS = 1;

#define BM 128
#define BN 128
#define KT 32

template<int OPB, int BIM, int REM, int STM>
__global__ __launch_bounds__(256)
void gemm_bf16_kernel(const bf16_t* __restrict__ A, long long sA, int lda,
                      const bf16_t* __restrict__ Bm, long long sB, int ldb,
                      float* D, long long sD, int ldd,
                      const float* __restrict__ bias,
                      const float* Res, long long sR, int ldr,
                      int M, int N, int K)
{
  __shared__ bf16_t lA[BM * LDSLD];
  __shared__ bf16_t lB[BN * LDSLD];

  const int tid  = threadIdx.x;
  const int lane = tid & 31;
  const int wid  = tid >> 5;
  const int wm   = (wid >> 2) * 64;   // wave row offset (2 rows of waves)
  const int wn   = (wid & 3) * 32;    // wave col offset (4 cols of waves)
  const long long bz = blockIdx.z;

  const bf16_t* Ab = A  + bz * sA;
  const bf16_t* Bb = Bm + bz * sB;
  float*        Db = D  + bz * sD;
  const float*  Rb = (REM == RE_NONE) ? nullptr : Res + bz * sR;

  const int m0 = blockIdx.y * BM;
  const int n0 = blockIdx.x * BN;

  const int row  = lane & 15;
  const int klo  = (lane & 16) ? 8 : 0;
  const int koff = (lane & 16) ? 16 : 0;

  v8f acc[4][2];
  v8f z8 = {0.f,0.f,0.f,0.f,0.f,0.f,0.f,0.f};
#pragma unroll
  for (int i = 0; i < 4; ++i)
#pragma unroll
    for (int j = 0; j < 2; ++j) acc[i][j] = z8;

  for (int kt = 0; kt < K; kt += KT) {
    // ---- stage A tile: rows m0..m0+127, k kt..kt+31 ----
    bool used_async = false;
    {
      int r  = tid >> 1;
      int kk = (tid & 1) * 16;
      int gm = m0 + r;
#if ASYNC_LDS
      if (gm < M) {
        const bf16_t* src = Ab + (long long)gm * lda + kt + kk;
        __builtin_prefetch(src + KT, 0, 0);
        __builtin_amdgcn_global_load_async_to_lds_b128(
            (gas_i4*)src,       (las_i4*)&lA[r * LDSLD + kk],     0, 0);
        __builtin_amdgcn_global_load_async_to_lds_b128(
            (gas_i4*)(src + 8), (las_i4*)&lA[r * LDSLD + kk + 8], 0, 0);
      }
      used_async = true;
#else
      uint4 d0 = {0,0,0,0}, d1 = {0,0,0,0};
      if (gm < M) {
        const bf16_t* src = Ab + (long long)gm * lda + kt + kk;
        __builtin_prefetch(src + KT, 0, 0);
        d0 = *(const uint4*)src;
        d1 = *(const uint4*)(src + 8);
      }
      *(uint4*)&lA[r * LDSLD + kk]     = d0;
      *(uint4*)&lA[r * LDSLD + kk + 8] = d1;
#endif
    }
    // ---- stage B tile as lB[n][k] ----
    if (OPB == OPB_T) {
      int r  = tid >> 1;
      int kk = (tid & 1) * 16;
      int gn = n0 + r;
#if ASYNC_LDS
      if (gn < N) {
        const bf16_t* src = Bb + (long long)gn * ldb + kt + kk;
        __builtin_prefetch(src + KT, 0, 0);
        __builtin_amdgcn_global_load_async_to_lds_b128(
            (gas_i4*)src,       (las_i4*)&lB[r * LDSLD + kk],     0, 0);
        __builtin_amdgcn_global_load_async_to_lds_b128(
            (gas_i4*)(src + 8), (las_i4*)&lB[r * LDSLD + kk + 8], 0, 0);
      }
#else
      uint4 d0 = {0,0,0,0}, d1 = {0,0,0,0};
      if (gn < N) {
        const bf16_t* src = Bb + (long long)gn * ldb + kt + kk;
        __builtin_prefetch(src + KT, 0, 0);
        d0 = *(const uint4*)src;
        d1 = *(const uint4*)(src + 8);
      }
      *(uint4*)&lB[r * LDSLD + kk]     = d0;
      *(uint4*)&lB[r * LDSLD + kk + 8] = d1;
#endif
    } else {
      int kk = tid >> 3;          // 0..31
      int nb = (tid & 7) * 16;    // 0..112
      int gn = n0 + nb;
      bf16_t tv[16];
      if (OPB == OPB_N) {
        const bf16_t* src = Bb + (long long)(kt + kk) * ldb + gn;
        uint4 d0 = *(const uint4*)src;
        uint4 d1 = *(const uint4*)(src + 8);
        *(uint4*)&tv[0] = d0;
        *(uint4*)&tv[8] = d1;
      } else { // OPB_C9 : B[k'][t] = Ypad[k'/9][t + k'%9]
        int kp  = kt + kk;
        int c   = kp / 9;
        int off = kp - c * 9;
        const bf16_t* src = Bb + (long long)c * ldb + gn + off;
#pragma unroll
        for (int j = 0; j < 16; ++j) tv[j] = src[j];
      }
#pragma unroll
      for (int j = 0; j < 16; ++j) lB[(nb + j) * LDSLD + kk] = tv[j];
    }
#if ASYNC_LDS
    if (used_async) __builtin_amdgcn_s_wait_asynccnt(0);
#else
    (void)used_async;
#endif
    __syncthreads();

    // ---- fragments + WMMA ----
    v16bf fa[4], fb[2];
#pragma unroll
    for (int mt = 0; mt < 4; ++mt)
      fa[mt] = lds_fragA(&lA[(wm + mt * 16 + row) * LDSLD], klo);
#pragma unroll
    for (int nt = 0; nt < 2; ++nt)
      fb[nt] = lds_fragB(&lB[(wn + nt * 16 + row) * LDSLD], klo);
#pragma unroll
    for (int mt = 0; mt < 4; ++mt)
#pragma unroll
      for (int nt = 0; nt < 2; ++nt)
        acc[mt][nt] = __builtin_amdgcn_wmma_f32_16x16x32_bf16(
            false, fa[mt], false, fb[nt], (short)0, acc[mt][nt], false, false);
    __syncthreads();
  }

  // ---- epilogue ----
#pragma unroll
  for (int mt = 0; mt < 4; ++mt)
#pragma unroll
    for (int nt = 0; nt < 2; ++nt)
#pragma unroll
      for (int r = 0; r < 8; ++r) {
        int gm = m0 + wm + mt * 16 + r + ((lane & 16) ? 8 : 0);
        int gn = n0 + wn + nt * 16 + (lane & 15);
        if (gm < M && gn < N) {
          float v = acc[mt][nt][r];
          if (BIM == BI_ROW) v += bias[gm];
          if (BIM == BI_COL) v += bias[gn];
          if (REM == RE_NORM)  v += Rb[(long long)gm * ldr + gn];
          if (REM == RE_TRANS) v += Rb[(long long)gn * ldr + gm];
          if (STM == STM_NORM) Db[(long long)gm * ldd + gn] = v;
          else                 Db[(long long)gn * ldd + gm] = v;
        }
      }
}

// ---------------------------------------------------------------------------
// Flash attention (wave32, WMMA bf16).  One wave per 16-query tile.
// Q,K,V,Out: fp32, [B, rows, 768], head h at cols h*64.
// ---------------------------------------------------------------------------
__global__ __launch_bounds__(128)
void attn_kernel(const float* __restrict__ Q, const float* __restrict__ K,
                 const float* __restrict__ V, float* __restrict__ Out,
                 int N, int kvN, int kvlen, int ldq, int hd)
{
  __shared__ bf16_t lp[4][16 * LDSLD];

  const int tid  = threadIdx.x;
  const int lane = tid & 31;
  const int wid  = tid >> 5;
  const int h = blockIdx.y, b = blockIdx.z;
  const int q0 = (blockIdx.x * 4 + wid) * 16;

  const float* Qb = Q + ((long long)b * N)   * ldq + h * hd;
  const float* Kb = K + ((long long)b * kvN) * ldq + h * hd;
  const float* Vb = V + ((long long)b * kvN) * ldq + h * hd;

  const int row  = lane & 15;
  const int klo  = (lane & 16) ? 8 : 0;
  const int koff = (lane & 16) ? 16 : 0;
  bf16_t* lpw = lp[wid];

  // Q fragments (A layout), pre-scaled by d^-1/2 = 1/8
  v16bf qf[2];
  {
    const float qs = 0.125f;
#pragma unroll
    for (int ks = 0; ks < 2; ++ks) {
      const float* qp = Qb + (long long)(q0 + row) * ldq + ks * 32 + klo;
      float t[16];
      *(float4*)&t[0]  = *(const float4*)(qp);
      *(float4*)&t[4]  = *(const float4*)(qp + 4);
      *(float4*)&t[8]  = *(const float4*)(qp + 16);
      *(float4*)&t[12] = *(const float4*)(qp + 20);
#pragma unroll
      for (int i = 0; i < 16; ++i) qf[ks][i] = f2bf(t[i] * qs);
    }
  }

  v8f z8 = {0.f,0.f,0.f,0.f,0.f,0.f,0.f,0.f};
  v8f o[4];
#pragma unroll
  for (int i = 0; i < 4; ++i) o[i] = z8;
  float mrow[8], lrow[8];
#pragma unroll
  for (int r = 0; r < 8; ++r) { mrow[r] = -3.0e38f; lrow[r] = 0.f; }

  const int jsteps = (kvlen + 31) / 32;
  for (int jt = 0; jt < jsteps; ++jt) {
    const int j = jt * 32;

    // --- S = Q * K^T for two 16-key subtiles ---
    v8f s0 = z8, s1 = z8;
#pragma unroll
    for (int st = 0; st < 2; ++st) {
      int tok  = j + st * 16 + row;
      int tokc = tok < kvlen ? tok : kvlen - 1;
#pragma unroll
      for (int ks = 0; ks < 2; ++ks) {
        const float* kp = Kb + (long long)tokc * ldq + ks * 32 + klo;
        float t[16];
        *(float4*)&t[0]  = *(const float4*)(kp);
        *(float4*)&t[4]  = *(const float4*)(kp + 4);
        *(float4*)&t[8]  = *(const float4*)(kp + 16);
        *(float4*)&t[12] = *(const float4*)(kp + 20);
        v16bf kf;
#pragma unroll
        for (int i = 0; i < 16; ++i) kf[i] = f2bf(t[i]);
        if (st == 0) s0 = __builtin_amdgcn_wmma_f32_16x16x32_bf16(false, qf[ks], false, kf, (short)0, s0, false, false);
        else         s1 = __builtin_amdgcn_wmma_f32_16x16x32_bf16(false, qf[ks], false, kf, (short)0, s1, false, false);
      }
    }
    // column-validity masking (per-lane uniform)
    bool v0 = (j + (lane & 15))      < kvlen;
    bool v1 = (j + 16 + (lane & 15)) < kvlen;
    if (!v0) {
#pragma unroll
      for (int r = 0; r < 8; ++r) s0[r] = -3.0e38f;
    }
    if (!v1) {
#pragma unroll
      for (int r = 0; r < 8; ++r) s1[r] = -3.0e38f;
    }

    // --- online softmax (half-wave reductions) ---
#pragma unroll
    for (int r = 0; r < 8; ++r) {
      float mx = fmaxf(s0[r], s1[r]);
#pragma unroll
      for (int msk = 1; msk < 16; msk <<= 1) mx = fmaxf(mx, __shfl_xor(mx, msk, 32));
      float mnew = fmaxf(mrow[r], mx);
      float sc = __expf(mrow[r] - mnew);
      float p0 = __expf(s0[r] - mnew);
      float p1 = __expf(s1[r] - mnew);
      float rs = p0 + p1;
#pragma unroll
      for (int msk = 1; msk < 16; msk <<= 1) rs += __shfl_xor(rs, msk, 32);
      lrow[r] = lrow[r] * sc + rs;
      mrow[r] = mnew;
      o[0][r] *= sc; o[1][r] *= sc; o[2][r] *= sc; o[3][r] *= sc;
      int prow = r + ((lane & 16) ? 8 : 0);
      lpw[prow * LDSLD + (lane & 15)]      = f2bf(p0);
      lpw[prow * LDSLD + 16 + (lane & 15)] = f2bf(p1);
    }

    // P fragment (A layout) from LDS (same-wave LDS ops are in-order)
    v16bf pa;
#pragma unroll
    for (int i = 0; i < 16; ++i) {
      int k = koff + i;
      pa[i] = lpw[row * LDSLD + k];
    }

    // --- O += P * V ---
#pragma unroll
    for (int nt = 0; nt < 4; ++nt) {
      v16bf vf;
      int dcol = nt * 16 + row;
#pragma unroll
      for (int i = 0; i < 16; ++i) {
        int tok  = j + koff + i;
        int tokc = tok < kvlen ? tok : kvlen - 1;
        vf[i] = f2bf(Vb[(long long)tokc * ldq + dcol]);
      }
      o[nt] = __builtin_amdgcn_wmma_f32_16x16x32_bf16(false, pa, false, vf, (short)0, o[nt], false, false);
    }
  }

  // --- store ---
#pragma unroll
  for (int nt = 0; nt < 4; ++nt)
#pragma unroll
    for (int r = 0; r < 8; ++r) {
      int gm = q0 + r + ((lane & 16) ? 8 : 0);
      int gn = h * hd + nt * 16 + (lane & 15);
      Out[((long long)b * N + gm) * ldq + gn] = o[nt][r] / lrow[r];
    }
}

// ---------------------------------------------------------------------------
// GroupNorm: x [B,C,T] fp32 -> token-major bf16 [B,T,C]
// ---------------------------------------------------------------------------
__global__ __launch_bounds__(256)
void groupnorm_kernel(const float* __restrict__ x, const float* __restrict__ sc,
                      const float* __restrict__ bi, bf16_t* __restrict__ out_t,
                      int C, int T, int cpg)
{
  int g = blockIdx.x, b = blockIdx.y;
  long long base = ((long long)b * C + (long long)g * cpg) * T;
  long long n = (long long)cpg * T;
  float s = 0.f, ss = 0.f;
  for (long long i = threadIdx.x; i < n; i += 256) {
    float v = x[base + i]; s += v; ss += v * v;
  }
  __shared__ float rs[256], rq[256];
  rs[threadIdx.x] = s; rq[threadIdx.x] = ss;
  __syncthreads();
  for (int off = 128; off > 0; off >>= 1) {
    if ((int)threadIdx.x < off) { rs[threadIdx.x] += rs[threadIdx.x + off]; rq[threadIdx.x] += rq[threadIdx.x + off]; }
    __syncthreads();
  }
  float mu   = rs[0] / (float)n;
  float var  = rq[0] / (float)n - mu * mu;
  float rstd = rsqrtf(var + 1e-6f);
  for (long long i = threadIdx.x; i < n; i += 256) {
    int c = g * cpg + (int)(i / T);
    int t = (int)(i % T);
    float v = (x[base + i] - mu) * rstd * sc[c] + bi[c];
    out_t[((long long)b * T + t) * C + c] = f2bf(v);
  }
}

// ---------------------------------------------------------------------------
// LayerNorm per token: h [Rows, I] fp32 -> bf16 (flat, or transposed+padded)
// ---------------------------------------------------------------------------
template<int TRANS>
__global__ __launch_bounds__(256)
void layernorm_kernel(const float* __restrict__ h, const float* __restrict__ sc,
                      const float* __restrict__ bi, bf16_t* __restrict__ out,
                      int I, int T, int Tpad)
{
  int rowid = blockIdx.x;
  const float* hp = h + (long long)rowid * I;
  float s = 0.f, ss = 0.f;
  for (int i = threadIdx.x; i < I; i += 256) { float v = hp[i]; s += v; ss += v * v; }
  __shared__ float rs[256], rq[256];
  rs[threadIdx.x] = s; rq[threadIdx.x] = ss;
  __syncthreads();
  for (int off = 128; off > 0; off >>= 1) {
    if ((int)threadIdx.x < off) { rs[threadIdx.x] += rs[threadIdx.x + off]; rq[threadIdx.x] += rq[threadIdx.x + off]; }
    __syncthreads();
  }
  float mu   = rs[0] / (float)I;
  float var  = rq[0] / (float)I - mu * mu;
  float rstd = rsqrtf(var + 1e-5f);
  for (int i = threadIdx.x; i < I; i += 256) {
    float v = (hp[i] - mu) * rstd * sc[i] + bi[i];
    if (!TRANS) {
      out[(long long)rowid * I + i] = f2bf(v);
    } else {
      int b = rowid / T, t = rowid % T;
      out[((long long)b * I + i) * Tpad + 4 + t] = f2bf(v);
    }
  }
}

// ---------------------------------------------------------------------------
// elementwise: fp32 -> bf16 ; GELU + transpose-pad
// ---------------------------------------------------------------------------
__global__ void cvt_f32_bf16(const float* __restrict__ in, bf16_t* __restrict__ out, long long n)
{
  long long i = (long long)blockIdx.x * 256 + threadIdx.x;
  if (i < n) out[i] = f2bf(in[i]);
}

__global__ void gelu_pad_kernel(const float* __restrict__ in, bf16_t* __restrict__ out,
                                long long n, int T, int Tpad)
{
  long long i = (long long)blockIdx.x * 256 + threadIdx.x;
  if (i < n) {
    long long bc = i / T;
    int t = (int)(i - bc * T);
    float xv = in[i];
    float g = 0.5f * xv * (1.0f + erff(xv * 0.70710678118654752f));
    out[bc * Tpad + 4 + t] = f2bf(g);
  }
}

// ---------------------------------------------------------------------------
// host orchestration
// ---------------------------------------------------------------------------
extern "C" void kernel_launch(void* const* d_in, const int* in_sizes, int n_in,
                              void* d_out, int out_size, void* d_ws, size_t ws_size,
                              hipStream_t stream)
{
  (void)in_sizes; (void)n_in; (void)out_size; (void)ws_size;
  constexpr int Bc = 2, Cc = 768, Tt = 2048, Ii = 768, Hh = 12, HD = 64;
  constexpr int CTXc = 512, Sc = 16, FI = 3072, KSc = 9, DEP = 2, NG = 32;
  constexpr int Tpad = Tt + 8;

  const float* x      = (const float*)d_in[0];
  const float* ctx    = (const float*)d_in[1];
  const float* gn_s   = (const float*)d_in[2];
  const float* gn_b   = (const float*)d_in[3];
  const float* pin_w  = (const float*)d_in[4];
  const float* pin_b  = (const float*)d_in[5];
  const float* n1_s   = (const float*)d_in[6];
  const float* n1_b   = (const float*)d_in[7];
  const float* a1_wq  = (const float*)d_in[8];
  const float* a1_wk  = (const float*)d_in[9];
  const float* a1_wv  = (const float*)d_in[10];
  const float* a1_wo  = (const float*)d_in[11];
  const float* a1_bo  = (const float*)d_in[12];
  const float* n2_s   = (const float*)d_in[13];
  const float* n2_b   = (const float*)d_in[14];
  const float* a2_wq  = (const float*)d_in[15];
  const float* a2_wk  = (const float*)d_in[16];
  const float* a2_wv  = (const float*)d_in[17];
  const float* a2_wo  = (const float*)d_in[18];
  const float* a2_bo  = (const float*)d_in[19];
  const float* n3_s   = (const float*)d_in[20];
  const float* n3_b   = (const float*)d_in[21];
  const float* ff1_w  = (const float*)d_in[22];
  const float* ff1_b  = (const float*)d_in[23];
  const float* ff2_w  = (const float*)d_in[24];
  const float* ff2_b  = (const float*)d_in[25];
  const float* pout_w = (const float*)d_in[26];
  const float* pout_b = (const float*)d_in[27];
  float* out = (float*)d_out;

  // workspace carve
  char* wp = (char*)d_ws;
  auto carve = [&](size_t bytes) -> void* {
    void* p = (void*)wp; wp += (bytes + 255) & ~((size_t)255); return p;
  };
  const size_t NTOK = (size_t)Bc * Tt;       // 4096 tokens
  float*  hbuf = (float*) carve(sizeof(float) * NTOK * Ii);
  float*  qbuf = (float*) carve(sizeof(float) * NTOK * Ii);
  float*  kbuf = (float*) carve(sizeof(float) * NTOK * Ii);
  float*  vbuf = (float*) carve(sizeof(float) * NTOK * Ii);
  float*  abuf = (float*) carve(sizeof(float) * NTOK * Ii);
  float*  gbuf = (float*) carve(sizeof(float) * (size_t)Bc * FI * Tt);
  bf16_t* ybf  = (bf16_t*)carve(2 * NTOK * Ii);
  bf16_t* y3p  = (bf16_t*)carve(2 * (size_t)Bc * Ii * Tpad);
  bf16_t* gpad = (bf16_t*)carve(2 * (size_t)Bc * FI * Tpad);
  bf16_t* wbig = (bf16_t*)carve(2 * (size_t)FI * Ii * KSc);
  bf16_t* wsm  = (bf16_t*)carve(2 * (size_t)Ii * Ii);
  bf16_t* ctxb = (bf16_t*)carve(2 * (size_t)Bc * Sc * CTXc);

#define CVT(src, dst, n) cvt_f32_bf16<<<divup_u((long long)(n), 256), 256, 0, stream>>>((src), (dst), (long long)(n))

  const dim3 gTok(Ii / BN, divup_u((long long)NTOK, BM), 1);   // token-major 4096 x 768

  // ---- GroupNorm -> ybf (token-major bf16) ----
  groupnorm_kernel<<<dim3(NG, Bc), 256, 0, stream>>>(x, gn_s, gn_b, ybf, Cc, Tt, Cc / NG);

  // ---- proj_in: h[t,o] = gn[t,:] . pin_w[o,:] + pin_b[o] ----
  CVT(pin_w, wsm, (size_t)Ii * Cc);
  gemm_bf16_kernel<OPB_T, BI_COL, RE_NONE, STM_NORM><<<gTok, 256, 0, stream>>>(
      ybf, 0, Cc, wsm, 0, Cc, hbuf, 0, Ii, pin_b, nullptr, 0, 0, (int)NTOK, Ii, Cc);

  // ---- context -> bf16 (once) ----
  CVT(ctx, ctxb, (size_t)Bc * Sc * CTXc);

  for (int d = 0; d < DEP; ++d) {
    const long long wII = (long long)d * Ii * Ii;
    // ===== self attention =====
    layernorm_kernel<0><<<(unsigned)NTOK, 256, 0, stream>>>(hbuf, n1_s + d * Ii, n1_b + d * Ii, ybf, Ii, Tt, 0);

    CVT(a1_wq + wII, wsm, (size_t)Ii * Ii);
    gemm_bf16_kernel<OPB_N, BI_NONE, RE_NONE, STM_NORM><<<gTok, 256, 0, stream>>>(
        ybf, 0, Ii, wsm, 0, Ii, qbuf, 0, Ii, nullptr, nullptr, 0, 0, (int)NTOK, Ii, Ii);
    CVT(a1_wk + wII, wsm, (size_t)Ii * Ii);
    gemm_bf16_kernel<OPB_N, BI_NONE, RE_NONE, STM_NORM><<<gTok, 256, 0, stream>>>(
        ybf, 0, Ii, wsm, 0, Ii, kbuf, 0, Ii, nullptr, nullptr, 0, 0, (int)NTOK, Ii, Ii);
    CVT(a1_wv + wII, wsm, (size_t)Ii * Ii);
    gemm_bf16_kernel<OPB_N, BI_NONE, RE_NONE, STM_NORM><<<gTok, 256, 0, stream>>>(
        ybf, 0, Ii, wsm, 0, Ii, vbuf, 0, Ii, nullptr, nullptr, 0, 0, (int)NTOK, Ii, Ii);

    attn_kernel<<<dim3(Tt / 64, Hh, Bc), 128, 0, stream>>>(qbuf, kbuf, vbuf, abuf, Tt, Tt, Tt, Ii, HD);

    CVT(abuf, ybf, NTOK * Ii);
    CVT(a1_wo + wII, wsm, (size_t)Ii * Ii);
    gemm_bf16_kernel<OPB_N, BI_COL, RE_NORM, STM_NORM><<<gTok, 256, 0, stream>>>(
        ybf, 0, Ii, wsm, 0, Ii, hbuf, 0, Ii, a1_bo + d * Ii, hbuf, 0, Ii, (int)NTOK, Ii, Ii);

    // ===== cross attention =====
    layernorm_kernel<0><<<(unsigned)NTOK, 256, 0, stream>>>(hbuf, n2_s + d * Ii, n2_b + d * Ii, ybf, Ii, Tt, 0);

    CVT(a2_wq + wII, wsm, (size_t)Ii * Ii);
    gemm_bf16_kernel<OPB_N, BI_NONE, RE_NONE, STM_NORM><<<gTok, 256, 0, stream>>>(
        ybf, 0, Ii, wsm, 0, Ii, qbuf, 0, Ii, nullptr, nullptr, 0, 0, (int)NTOK, Ii, Ii);
    CVT(a2_wk + (long long)d * CTXc * Ii, wsm, (size_t)CTXc * Ii);
    gemm_bf16_kernel<OPB_N, BI_NONE, RE_NONE, STM_NORM><<<dim3(Ii / BN, 1, Bc), 256, 0, stream>>>(
        ctxb, (long long)Sc * CTXc, CTXc, wsm, 0, Ii, kbuf, (long long)Sc * Ii, Ii,
        nullptr, nullptr, 0, 0, Sc, Ii, CTXc);
    CVT(a2_wv + (long long)d * CTXc * Ii, wsm, (size_t)CTXc * Ii);
    gemm_bf16_kernel<OPB_N, BI_NONE, RE_NONE, STM_NORM><<<dim3(Ii / BN, 1, Bc), 256, 0, stream>>>(
        ctxb, (long long)Sc * CTXc, CTXc, wsm, 0, Ii, vbuf, (long long)Sc * Ii, Ii,
        nullptr, nullptr, 0, 0, Sc, Ii, CTXc);

    attn_kernel<<<dim3(Tt / 64, Hh, Bc), 128, 0, stream>>>(qbuf, kbuf, vbuf, abuf, Tt, Sc, Sc, Ii, HD);

    CVT(abuf, ybf, NTOK * Ii);
    CVT(a2_wo + wII, wsm, (size_t)Ii * Ii);
    gemm_bf16_kernel<OPB_N, BI_COL, RE_NORM, STM_NORM><<<gTok, 256, 0, stream>>>(
        ybf, 0, Ii, wsm, 0, Ii, hbuf, 0, Ii, a2_bo + d * Ii, hbuf, 0, Ii, (int)NTOK, Ii, Ii);

    // ===== conv feed-forward =====
    (void)hipMemsetAsync(y3p, 0, 2 * (size_t)Bc * Ii * Tpad, stream);
    layernorm_kernel<1><<<(unsigned)NTOK, 256, 0, stream>>>(hbuf, n3_s + d * Ii, n3_b + d * Ii, y3p, Ii, Tt, Tpad);

    CVT(ff1_w + (long long)d * FI * Ii * KSc, wbig, (size_t)FI * Ii * KSc);
    gemm_bf16_kernel<OPB_C9, BI_ROW, RE_NONE, STM_NORM><<<dim3(Tt / BN, FI / BM, Bc), 256, 0, stream>>>(
        wbig, 0, Ii * KSc, y3p, (long long)Ii * Tpad, Tpad,
        gbuf, (long long)FI * Tt, Tt, ff1_b + d * FI, nullptr, 0, 0, FI, Tt, Ii * KSc);

    (void)hipMemsetAsync(gpad, 0, 2 * (size_t)Bc * FI * Tpad, stream);
    gelu_pad_kernel<<<divup_u((long long)Bc * FI * Tt, 256), 256, 0, stream>>>(
        gbuf, gpad, (long long)Bc * FI * Tt, Tt, Tpad);

    CVT(ff2_w + (long long)d * Ii * FI * KSc, wbig, (size_t)Ii * FI * KSc);
    gemm_bf16_kernel<OPB_C9, BI_ROW, RE_TRANS, STM_TRANS><<<dim3(Tt / BN, Ii / BM, Bc), 256, 0, stream>>>(
        wbig, 0, FI * KSc, gpad, (long long)FI * Tpad, Tpad,
        hbuf, (long long)Tt * Ii, Ii, ff2_b + d * Ii, hbuf, (long long)Tt * Ii, Ii, Ii, Tt, FI * KSc);
  }

  // ---- proj_out + input residual:  out[b,o,t] = h[b,t,:].pout_w[o,:] + pout_b[o] + x[b,o,t]
  CVT(hbuf, ybf, NTOK * Ii);
  CVT(pout_w, wsm, (size_t)Cc * Ii);
  gemm_bf16_kernel<OPB_T, BI_COL, RE_TRANS, STM_TRANS><<<dim3(Cc / BN, Tt / BM, Bc), 256, 0, stream>>>(
      ybf, (long long)Tt * Ii, Ii, wsm, 0, Ii, out, (long long)Cc * Tt, Tt,
      pout_b, x, (long long)Cc * Tt, Tt, Tt, Cc, Ii);

#undef CVT
}
